// Block_1838246002769
// MI455X (gfx1250) — matrix operand order repaired and checked by reading, not compiled, not run on hardware
//
#include <hip/hip_runtime.h>

typedef unsigned short u16;
typedef __attribute__((ext_vector_type(16))) __bf16 v16bf;
typedef __attribute__((ext_vector_type(8)))  float  v8f;

union AFrag { float4 f[2]; v16bf v; };

__device__ __forceinline__ v8f wmma_bf16(v16bf a, v16bf b, v8f c) {
  // D = A(16x32 bf16) * B(32x16 bf16) + C(16x16 f32)
  return __builtin_amdgcn_wmma_f32_16x16x32_bf16(false, a, false, b, (short)0, c,
                                                 false, false);
}

__device__ __forceinline__ u16 f2bf(float x) {
  unsigned u = __float_as_uint(x);
  return (u16)((u + 0x7FFFu + ((u >> 16) & 1u)) >> 16);
}

// LDS byte offset of a generic pointer to __shared__ (flat LDS aperture: low 32 bits = offset)
__device__ __forceinline__ unsigned lds_off(const void* p) {
  return (unsigned)(size_t)p;
}

// CDNA5 async global->LDS copy, 16 bytes per lane, tracked by ASYNCcnt
__device__ __forceinline__ void async_copy_b128(unsigned loff, const u16* g) {
  asm volatile("global_load_async_to_lds_b128 %0, %1, off"
               :: "v"(loff), "v"(g) : "memory");
}
__device__ __forceinline__ void wait_async0() {
  asm volatile("s_wait_asynccnt 0x0" ::: "memory");
}

// ---------------------------------------------------------------- tiled cast+transpose f32 -> bf16
// dst[n*Kd + k] = bf16(src[k*ld + off + n]); grid = (N/32, Kd/32), block = 256
__global__ __launch_bounds__(256) void cast_transpose(const float* __restrict__ src,
                                                      u16* __restrict__ dst,
                                                      int Kd, int ld, int off) {
  __shared__ u16 tile[32][33];
  int kb = blockIdx.y * 32, nb = blockIdx.x * 32;
  int tx = threadIdx.x & 31, ty = threadIdx.x >> 5;
#pragma unroll
  for (int r = ty; r < 32; r += 8)
    tile[r][tx] = f2bf(src[(size_t)(kb + r) * ld + off + nb + tx]);  // coalesced in n
  __syncthreads();
#pragma unroll
  for (int r = ty; r < 32; r += 8)
    dst[(size_t)(nb + r) * Kd + kb + tx] = tile[tx][r];              // coalesced in k
}

// ---------------------------------------------------------------- LayerNorm (D=1024) -> bf16
__global__ __launch_bounds__(256) void ln_kernel(const float* __restrict__ x,
                                                 const float* __restrict__ g,
                                                 const float* __restrict__ b,
                                                 u16* __restrict__ out) {
  __shared__ float2 red[256];
  const int D = 1024;
  int row = blockIdx.x, tid = threadIdx.x;
  const float* xr = x + (size_t)row * D;
  float4 v = ((const float4*)xr)[tid];
  float s  = v.x + v.y + v.z + v.w;
  float s2 = v.x * v.x + v.y * v.y + v.z * v.z + v.w * v.w;
  red[tid] = make_float2(s, s2);
  __syncthreads();
  for (int st = 128; st > 0; st >>= 1) {
    if (tid < st) {
      red[tid].x += red[tid + st].x;
      red[tid].y += red[tid + st].y;
    }
    __syncthreads();
  }
  float mu  = red[0].x * (1.0f / D);
  float var = red[0].y * (1.0f / D) - mu * mu;
  float rs  = rsqrtf(var + 1e-5f);
  float4 gv = ((const float4*)g)[tid];
  float4 bv = ((const float4*)b)[tid];
  u16 o4[4];
  o4[0] = f2bf((v.x - mu) * rs * gv.x + bv.x);
  o4[1] = f2bf((v.y - mu) * rs * gv.y + bv.y);
  o4[2] = f2bf((v.z - mu) * rs * gv.z + bv.z);
  o4[3] = f2bf((v.w - mu) * rs * gv.w + bv.w);
  *(uint2*)(out + (size_t)row * D + tid * 4) = *(uint2*)o4;
}

// ---------------------------------------------------------------- bf16 WMMA GEMM
// (double-buffered, async global->LDS staging)
// C(MxN,f32 acc) = A(MxK bf16 rowmajor) * Bt(NxK bf16, pre-transposed) + bias.
#define MODE_F32_RESID 0
#define MODE_BF16_RELU 1
#define MODE_KSCATTER  2
#define BM 128
#define BN 128
#define BK 32
#define LDSS 40   // padded stride in bf16 units (80B: 16B-aligned, spreads banks)

__global__ __launch_bounds__(256) void gemm_bf16(const u16* __restrict__ A,
                                                 const u16* __restrict__ Bt,
                                                 const float* __restrict__ bias,
                                                 const float* __restrict__ resid,
                                                 void* __restrict__ outp,
                                                 u16* __restrict__ out2,
                                                 int M, int N, int K, int mode) {
  __shared__ u16 sA[2][BM * LDSS];   // A tile [m][k]
  __shared__ u16 sB[2][BN * LDSS];   // B tile [n][k]
  int tid = threadIdx.x;
  int wave = tid >> 5, lane = tid & 31;
  int wm = wave >> 1, wn = wave & 1;        // 4x2 wave grid -> 32x64 per wave
  int lh = lane >> 4, ll = lane & 15;
  int m0 = blockIdx.y * BM, n0 = blockIdx.x * BN;

  int srow = tid >> 1, skc = (tid & 1) * 16;      // staging: 32B per thread per tile
  const u16* gA = A  + (size_t)(m0 + srow) * K + skc;
  const u16* gB = Bt + (size_t)(n0 + srow) * K + skc;
  unsigned oA[2], oB[2];
  oA[0] = lds_off(&sA[0][srow * LDSS + skc]);
  oA[1] = lds_off(&sA[1][srow * LDSS + skc]);
  oB[0] = lds_off(&sB[0][srow * LDSS + skc]);
  oB[1] = lds_off(&sB[1][srow * LDSS + skc]);

  v8f acc[2][4];
#pragma unroll
  for (int i = 0; i < 2; ++i)
#pragma unroll
    for (int j = 0; j < 4; ++j) acc[i][j] = (v8f){};

  // preload tile 0 into buffer 0 (async direct-to-LDS)
  async_copy_b128(oA[0],      gA);
  async_copy_b128(oA[0] + 32, gA + 16);
  async_copy_b128(oB[0],      gB);
  async_copy_b128(oB[0] + 32, gB + 16);
  wait_async0();
  __syncthreads();

  int nIter = K / BK;
  for (int it = 0; it < nIter; ++it) {
    int buf = it & 1;
    // kick off next tile's async copies into the other buffer
    // (all waves finished reading it before the previous barrier)
    if (it + 1 < nIter) {
      const u16* nA = gA + (it + 1) * BK;
      const u16* nB = gB + (it + 1) * BK;
      async_copy_b128(oA[buf ^ 1],      nA);
      async_copy_b128(oA[buf ^ 1] + 32, nA + 16);
      async_copy_b128(oB[buf ^ 1],      nB);
      async_copy_b128(oB[buf ^ 1] + 32, nB + 16);
    }
    // compute current tile while the async copies are in flight
    AFrag af[2], bfr[4];
#pragma unroll
    for (int i = 0; i < 2; ++i) {
      int row = wm * 32 + i * 16 + ll;
      af[i].f[0] = *(const float4*)(sA[buf] + row * LDSS + lh * 8);
      af[i].f[1] = *(const float4*)(sA[buf] + row * LDSS + 16 + lh * 8);
    }
#pragma unroll
    for (int j = 0; j < 4; ++j) {
      int col = wn * 64 + j * 16 + ll;
      const u16* p = sB[buf] + col * LDSS + lh * 8;
      bfr[j].f[0] = *(const float4*)p;
      bfr[j].f[1] = *(const float4*)(p + 16);
    }
#pragma unroll
    for (int i = 0; i < 2; ++i)
#pragma unroll
      for (int j = 0; j < 4; ++j)
        acc[i][j] = wmma_bf16(af[i].v, bfr[j].v, acc[i][j]);
    wait_async0();      // our copies into buf^1 are done
    __syncthreads();    // everyone's copies are done / reads of buf finished
  }

  // epilogue
#pragma unroll
  for (int i = 0; i < 2; ++i) {
#pragma unroll
    for (int j = 0; j < 4; ++j) {
      int rbase = m0 + wm * 32 + i * 16 + 8 * lh;
      int col   = n0 + wn * 64 + j * 16 + ll;
      float bv = bias[col];
#pragma unroll
      for (int r = 0; r < 8; ++r) {
        int row = rbase + r;
        float val = acc[i][j][r] + bv;
        if (mode == MODE_F32_RESID) {
          ((float*)outp)[(size_t)row * N + col] = val + resid[(size_t)row * N + col];
        } else if (mode == MODE_BF16_RELU) {
          ((u16*)outp)[(size_t)row * N + col] = f2bf(fmaxf(val, 0.0f));
        } else {  // MODE_KSCATTER: row = b*2048+l, col = h*64+d (N==1024)
          int bb = row >> 11, l = row & 2047;
          int h = col >> 6, d = col & 63;
          u16 kv = f2bf(val);
          ((u16*)outp)[(((size_t)(bb * 16 + h) * 2048 + l) << 6) + d] = kv;  // K [bh][l][d]
          out2[(((size_t)(bb * 16 + h) * 64 + d) << 11) + l] = kv;           // Kt [bh][d][l]
        }
      }
    }
  }
}

// ---------------------------------------------------------------- flash-style K*K^T attention
// 1 wave = 16 queries; block = 4 waves; grid = (L/64, B*H)
__global__ __launch_bounds__(128) void attn_kernel(const u16* __restrict__ Kbf,
                                                   const u16* __restrict__ Kt,
                                                   u16* __restrict__ O) {
  __shared__ u16 sP[4][16 * LDSS];   // per-wave P staging (C-layout -> A-layout)
  int tid = threadIdx.x, wave = tid >> 5, lane = tid & 31;
  int lh = lane >> 4, ll = lane & 15;
  int bh = blockIdx.y;
  int q0 = blockIdx.x * 64 + wave * 16;
  const u16* Kb  = Kbf + (size_t)bh * 2048 * 64;
  const u16* Ktb = Kt  + (size_t)bh * 64 * 2048;

  // Q A-fragments (queries ARE K rows): 16x64 split as two 16x32 frags
  AFrag qf[2];
  {
    const u16* qrow = Kb + (size_t)(q0 + ll) * 64;
#pragma unroll
    for (int f = 0; f < 2; ++f) {
      qf[f].f[0] = *(const float4*)(qrow + f * 32 + lh * 8);
      qf[f].f[1] = *(const float4*)(qrow + f * 32 + 16 + lh * 8);
    }
  }

  v8f o[4];
#pragma unroll
  for (int f = 0; f < 4; ++f) o[f] = (v8f){};
  float mrow[8], lrow[8];
#pragma unroll
  for (int r = 0; r < 8; ++r) { mrow[r] = -1e30f; lrow[r] = 0.0f; }

  for (int kb = 0; kb < 2048; kb += 32) {
    if (kb + 32 < 2048) {  // WGP-scope prefetch of next key tile
      __builtin_prefetch(Kb + (size_t)(kb + 32 + lane) * 64, 0, 3);
      __builtin_prefetch(Ktb + (size_t)lane * 2048 + kb + 32, 0, 3);
    }
    // ---- scores: two 16-key subtiles, S = Q * K^T (B = Kbf rows directly, N-major)
    v8f s[2];
#pragma unroll
    for (int t = 0; t < 2; ++t) {
      const u16* krow = Kb + (size_t)(kb + t * 16 + ll) * 64;
      AFrag b0, b1;
      b0.f[0] = *(const float4*)(krow + lh * 8);
      b0.f[1] = *(const float4*)(krow + 16 + lh * 8);
      b1.f[0] = *(const float4*)(krow + 32 + lh * 8);
      b1.f[1] = *(const float4*)(krow + 48 + lh * 8);
      v8f z = (v8f){};
      z = wmma_bf16(qf[0].v, b0.v, z);
      z = wmma_bf16(qf[1].v, b1.v, z);
      s[t] = z;
    }

    // ---- online softmax per row (row r+8*lh lives in VGPR r across the 16-lane half)
#pragma unroll
    for (int r = 0; r < 8; ++r) {
      float a0 = s[0][r] * 0.125f, a1 = s[1][r] * 0.125f;  // 1/sqrt(64)
      float mx = fmaxf(a0, a1);
#pragma unroll
      for (int off = 1; off < 16; off <<= 1) mx = fmaxf(mx, __shfl_xor(mx, off, 32));
      float mnew  = fmaxf(mrow[r], mx);
      float alpha = __expf(mrow[r] - mnew);
      float p0 = __expf(a0 - mnew), p1 = __expf(a1 - mnew);
      float rsum = p0 + p1;
#pragma unroll
      for (int off = 1; off < 16; off <<= 1) rsum += __shfl_xor(rsum, off, 32);
      lrow[r] = lrow[r] * alpha + rsum;
      mrow[r] = mnew;
#pragma unroll
      for (int f = 0; f < 4; ++f) o[f][r] *= alpha;
      int prow = (r + 8 * lh) * LDSS;
      sP[wave][prow + ll]      = f2bf(p0);
      sP[wave][prow + 16 + ll] = f2bf(p1);
    }
    asm volatile("s_wait_dscnt 0" ::: "memory");  // LDS round-trip (wave-private)

    // ---- P as A-fragment (16x32), PV against Kt (N-major B)
    AFrag pa;
    pa.f[0] = *(const float4*)(&sP[wave][ll * LDSS + lh * 8]);
    pa.f[1] = *(const float4*)(&sP[wave][ll * LDSS + 16 + lh * 8]);
#pragma unroll
    for (int f = 0; f < 4; ++f) {
      const u16* kt = Ktb + (size_t)(f * 16 + ll) * 2048 + kb + lh * 8;
      AFrag vb;
      vb.f[0] = *(const float4*)kt;
      vb.f[1] = *(const float4*)(kt + 16);
      o[f] = wmma_bf16(pa.v, vb.v, o[f]);
    }
  }

  // ---- normalize + write O[b][q][h*64+d] (bf16)
  int bb = bh >> 4, h = bh & 15;
#pragma unroll
  for (int f = 0; f < 4; ++f) {
#pragma unroll
    for (int r = 0; r < 8; ++r) {
      int q = q0 + r + 8 * lh;
      float val = o[f][r] / lrow[r];
      O[((size_t)(bb * 2048 + q)) * 1024 + h * 64 + f * 16 + ll] = f2bf(val);
    }
  }
}

// ---------------------------------------------------------------- host
extern "C" void kernel_launch(void* const* d_in, const int* in_sizes, int n_in,
                              void* d_out, int out_size, void* d_ws, size_t ws_size,
                              hipStream_t stream) {
  const float* x      = (const float*)d_in[0];
  const float* w_attn = (const float*)d_in[1];
  const float* b_attn = (const float*)d_in[2];
  const float* w_proj = (const float*)d_in[3];
  const float* b_proj = (const float*)d_in[4];
  const float* ln1_g  = (const float*)d_in[5];
  const float* ln1_b  = (const float*)d_in[6];
  const float* ln2_g  = (const float*)d_in[7];
  const float* ln2_b  = (const float*)d_in[8];
  const float* w_fc1  = (const float*)d_in[9];
  const float* b_fc1  = (const float*)d_in[10];
  const float* w_fc2  = (const float*)d_in[11];
  const float* b_fc2  = (const float*)d_in[12];

  const int D = 1024, M = 4096;  // M = B*L rows

  char* ws = (char*)d_ws;
  size_t off = 0;
  auto alloc = [&](size_t bytes) -> void* {
    void* p = ws + off;
    off += (bytes + 255) & ~(size_t)255;
    return p;
  };
  // weights stored TRANSPOSED: wt[n*K + k]
  u16* wk    = (u16*)alloc((size_t)1024 * 1024 * 2);   // [1024 n][1024 k]
  u16* wpj   = (u16*)alloc((size_t)1024 * 1024 * 2);
  u16* wf1   = (u16*)alloc((size_t)4096 * 1024 * 2);   // [4096 n][1024 k]
  u16* wf2   = (u16*)alloc((size_t)1024 * 4096 * 2);   // [1024 n][4096 k]
  u16* hbf   = (u16*)alloc((size_t)M * 1024 * 2);
  u16* Kbf   = (u16*)alloc((size_t)M * 1024 * 2);
  u16* Ktb   = (u16*)alloc((size_t)M * 1024 * 2);
  u16* Obf   = (u16*)alloc((size_t)M * 1024 * 2);
  float* x1  = (float*)alloc((size_t)M * 1024 * 4);
  u16* f1bf  = (u16*)alloc((size_t)M * 4096 * 2);

  // 1) weight cast+transpose (K-slice of w_attn; w_proj; w_fc1; w_fc2)
  cast_transpose<<<dim3(1024 / 32, 1024 / 32), 256, 0, stream>>>(w_attn, wk, 1024, 3072, 1024);
  cast_transpose<<<dim3(1024 / 32, 1024 / 32), 256, 0, stream>>>(w_proj, wpj, 1024, 1024, 0);
  cast_transpose<<<dim3(4096 / 32, 1024 / 32), 256, 0, stream>>>(w_fc1, wf1, 1024, 4096, 0);
  cast_transpose<<<dim3(1024 / 32, 4096 / 32), 256, 0, stream>>>(w_fc2, wf2, 4096, 1024, 0);

  // 2) h = LN1(x)
  ln_kernel<<<M, 256, 0, stream>>>(x, ln1_g, ln1_b, hbf);

  // 3) K = h @ w_attn[:,D:2D] + b_attn[D:2D], scattered to [bh][l][d] and [bh][d][l]
  gemm_bf16<<<dim3(1024 / BN, M / BM), 256, 0, stream>>>(
      hbf, wk, b_attn + D, nullptr, Kbf, Ktb, M, 1024, 1024, MODE_KSCATTER);

  // 4) attention (K as Q,K,V)
  attn_kernel<<<dim3(2048 / 64, 32), 128, 0, stream>>>(Kbf, Ktb, Obf);

  // 5) x1 = x + O @ w_proj + b_proj
  gemm_bf16<<<dim3(1024 / BN, M / BM), 256, 0, stream>>>(
      Obf, wpj, b_proj, x, x1, nullptr, M, 1024, 1024, MODE_F32_RESID);

  // 6) h = LN2(x1)
  ln_kernel<<<M, 256, 0, stream>>>(x1, ln2_g, ln2_b, hbf);

  // 7) f1 = relu(h @ w_fc1 + b_fc1)
  gemm_bf16<<<dim3(4096 / BN, M / BM), 256, 0, stream>>>(
      hbf, wf1, b_fc1, nullptr, f1bf, nullptr, M, 4096, 1024, MODE_BF16_RELU);

  // 8) out = x1 + f1 @ w_fc2 + b_fc2
  gemm_bf16<<<dim3(1024 / BN, M / BM), 256, 0, stream>>>(
      f1bf, wf2, b_fc2, x1, d_out, nullptr, M, 1024, 4096, MODE_F32_RESID);
}
